// Density_89790586290724
// MI455X (gfx1250) — matrix operand order repaired and checked by reading, not compiled, not run on hardware
//
#include <hip/hip_runtime.h>
#include <cstdint>
#include <cstddef>

// ---------------------------------------------------------------------------
// Instant-NGP hash-grid encode (L=24, T=2^19, F=2) + 3-layer MLP (48->64->64->33)
// CDNA5 / gfx1250: wave32, V_WMMA_F32_16X16X32_F16 for all three GEMMs.
// Each wave of 32 lanes owns 32 points (two 16-row WMMA tiles).
//  - LDS tiles padded to 72-half rows (36 dwords) -> conflict-free ds_load_b128
//  - outputs/points use non-temporal hints so the 100MB table stays L2-resident
// ---------------------------------------------------------------------------

typedef __attribute__((ext_vector_type(16))) _Float16 h16_t;  // WMMA A/B frag (8 VGPRs)
typedef __attribute__((ext_vector_type(8)))  _Float16 h8_t;   // 16-byte LDS unit
typedef __attribute__((ext_vector_type(8)))  float    f8_t;   // WMMA C/D frag (8 VGPRs)

#define NLVL 24
#define TMASK ((1u << 19) - 1u)
#define WAVES_PER_BLK 4
#define PTS_PER_WAVE 32
#define PTS_PER_BLK (WAVES_PER_BLK * PTS_PER_WAVE)
#define TSTRIDE 72   // halves per tile row (pad 64 -> 72: 36 dwords, bank-conflict-free)

// RES = floor(16 * (128^(1/23))^l), precomputed
__constant__ float kRes[NLVL] = {
    16.f, 19.f, 24.f, 30.f, 37.f, 45.f, 56.f, 70.f,
    86.f, 106.f, 131.f, 162.f, 201.f, 248.f, 306.f, 378.f,
    467.f, 577.f, 713.f, 880.f, 1087.f, 1343.f, 1658.f, 2048.f};

__device__ __forceinline__ h16_t cat16(h8_t lo, h8_t hi) {
  h16_t r;
#pragma unroll
  for (int i = 0; i < 8; ++i) { r[i] = lo[i]; r[i + 8] = hi[i]; }
  return r;
}

// A fragment (16x32 f16) from a row-major [32][TSTRIDE] f16 LDS tile.
// ISA layout: lane holds row M=lane%16; half h holds K = (h&8?16:0)+(laneHi?8:0)+(h&7).
__device__ __forceinline__ h16_t loadA(const _Float16* tile, int mt, int kt, int lane) {
  const int row = mt * 16 + (lane & 15);
  const int kb  = kt * 32 + ((lane >> 4) & 1) * 8;
  h8_t lo = *(const h8_t*)(tile + row * TSTRIDE + kb);        // ds_load_b128
  h8_t hi = *(const h8_t*)(tile + row * TSTRIDE + kb + 16);   // ds_load_b128
  return cat16(lo, hi);
}

// B fragment pre-swizzled in LDS: frag is 512 contiguous halves, [lane][16 halves].
__device__ __forceinline__ h16_t loadB(const _Float16* frag, int lane) {
  const h8_t* p = (const h8_t*)(frag + lane * 16);
  return cat16(p[0], p[1]);
}

// Stage W (row-major KxNn f32 in global) into LDS as f16 WMMA-B fragments.
// Fragment (nt,kt) -> dst[(nt*KT+kt)*512 ..]; within: lane L, half h maps to
// n = nt*16 + L%16 ; k = kt*32 + (h&8?16:0) + (L>=16?8:0) + (h&7)   (32x16 f16 B layout).
__device__ __forceinline__ void stageW(_Float16* dst, const float* W,
                                       int K, int Nn, int KT, int NT,
                                       int tid, int nthr) {
  const int total = KT * NT * 512;
  for (int idx = tid; idx < total; idx += nthr) {
    const int frag = idx >> 9, rem = idx & 511;
    const int L = rem >> 4, h = rem & 15;
    const int nt = frag / KT, kt = frag - nt * KT;
    const int n = nt * 16 + (L & 15);
    const int k = kt * 32 + ((h >> 3) << 4) + ((L >> 4) << 3) + (h & 7);
    const float v = (k < K && n < Nn) ? W[k * Nn + n] : 0.f;
    dst[idx] = (_Float16)v;
  }
}

__global__ void __launch_bounds__(PTS_PER_BLK)
ngp_density_kernel(const float* __restrict__ points, const float* __restrict__ bb,
                   const float* __restrict__ tbl,
                   const float* __restrict__ w1, const float* __restrict__ b1,
                   const float* __restrict__ w2, const float* __restrict__ b2,
                   const float* __restrict__ w3, const float* __restrict__ b3,
                   float* __restrict__ out, int N) {
  __shared__ __align__(16) _Float16 sEnc[WAVES_PER_BLK][32 * TSTRIDE]; // enc, reused as h2
  __shared__ __align__(16) _Float16 sH1 [WAVES_PER_BLK][32 * TSTRIDE]; // h1
  __shared__ __align__(16) _Float16 sW1[8 * 512];                      // 2 KT x 4 NT frags
  __shared__ __align__(16) _Float16 sW2[8 * 512];                      // 2 KT x 4 NT frags
  __shared__ __align__(16) _Float16 sW3[6 * 512];                      // 2 KT x 3 NT frags
  __shared__ float sB1[64], sB2[64], sB3[48];

  const int tid  = threadIdx.x;
  const int wv   = tid >> 5;
  const int lane = tid & 31;

  // ---- stage weights (f32 -> f16, B-fragment swizzle) ----------------------
  stageW(sW1, w1, 48, 64, 2, 4, tid, PTS_PER_BLK);
  stageW(sW2, w2, 64, 64, 2, 4, tid, PTS_PER_BLK);
  stageW(sW3, w3, 64, 33, 2, 3, tid, PTS_PER_BLK);
  if (tid < 64) { sB1[tid] = b1[tid]; sB2[tid] = b2[tid]; }
  if (tid < 48) { sB3[tid] = (tid < 33) ? b3[tid] : 0.f; }

  _Float16* enc = sEnc[wv];
  _Float16* h1  = sH1[wv];

  const int tileBase = blockIdx.x * PTS_PER_BLK + wv * PTS_PER_WAVE;
  const int p = tileBase + lane;

  // ---- hash-grid encoding: one point per lane ------------------------------
  float px = 0.f, py = 0.f, pz = 0.f;
  if (p < N) {  // streamed once: keep out of L2's way
    px = __builtin_nontemporal_load(points + p * 3 + 0);
    py = __builtin_nontemporal_load(points + p * 3 + 1);
    pz = __builtin_nontemporal_load(points + p * 3 + 2);
  }
  const float x = fminf(fmaxf(px / bb[0] + 0.5f, 0.f), 1.f);
  const float y = fminf(fmaxf(py / bb[1] + 0.5f, 0.f), 1.f);
  const float z = fminf(fmaxf(pz / bb[2] + 0.5f, 0.f), 1.f);

  // zero the K-padding columns 48..63 of this lane's enc row (two b128 stores)
  {
    h8_t zz = {};
    *(h8_t*)(enc + lane * TSTRIDE + 48) = zz;
    *(h8_t*)(enc + lane * TSTRIDE + 56) = zz;
  }

  const float2* tb = (const float2*)tbl;
  // 6 groups of 4 levels: 32 independent table gathers per group, one b128 store
#pragma unroll
  for (int lg = 0; lg < 6; ++lg) {
    h8_t pack;
#pragma unroll
    for (int li = 0; li < 4; ++li) {
      const int l = lg * 4 + li;
      const float res = kRes[l];
      const float fx = x * res, fy = y * res, fz = z * res;
      const float ixf = floorf(fx), iyf = floorf(fy), izf = floorf(fz);
      const float wx1 = fx - ixf, wy1 = fy - iyf, wz1 = fz - izf;
      const float wx0 = 1.f - wx1, wy0 = 1.f - wy1, wz0 = 1.f - wz1;
      const uint32_t ix = (uint32_t)ixf, iy = (uint32_t)iyf, iz = (uint32_t)izf;
      const uint32_t hx0 = ix,               hx1 = ix + 1u;
      const uint32_t hy0 = iy * 2654435761u, hy1 = (iy + 1u) * 2654435761u;
      const uint32_t hz0 = iz * 805459861u,  hz1 = (iz + 1u) * 805459861u;
      const float2* tl = tb + ((size_t)l << 19);
      float f0 = 0.f, f1 = 0.f;
#define CRN(bx, by, bz)                                                        \
  {                                                                            \
    const uint32_t hh = ((bx ? hx1 : hx0) ^ (by ? hy1 : hy0) ^                 \
                        (bz ? hz1 : hz0)) & TMASK;                             \
    const float2 ft = tl[hh];                                                  \
    const float wt = (bx ? wx1 : wx0) * (by ? wy1 : wy0) * (bz ? wz1 : wz0);   \
    f0 = fmaf(wt, ft.x, f0); f1 = fmaf(wt, ft.y, f1);                          \
  }
      CRN(0, 0, 0) CRN(1, 0, 0) CRN(0, 1, 0) CRN(1, 1, 0)
      CRN(0, 0, 1) CRN(1, 0, 1) CRN(0, 1, 1) CRN(1, 1, 1)
#undef CRN
      pack[2 * li]     = (_Float16)f0;
      pack[2 * li + 1] = (_Float16)f1;
    }
    *(h8_t*)(enc + lane * TSTRIDE + lg * 8) = pack;  // ds_store_b128
  }

  __syncthreads();  // weights staged; enc/h1 tiles are wave-private (DS in-order)

  const int nLoc = lane & 15;              // D-frag column within tile
  const int mOff = ((lane >> 4) & 1) * 8;  // D-frag row base (lanes 16-31 -> M+8)

  // ---- layer 1: enc(32x64) x W1 -> relu -> h1(32x64) -----------------------
#pragma unroll
  for (int mt = 0; mt < 2; ++mt) {
#pragma unroll
    for (int nt = 0; nt < 4; ++nt) {
      f8_t acc = {};
#pragma unroll
      for (int kt = 0; kt < 2; ++kt) {
        h16_t a = loadA(enc, mt, kt, lane);
        h16_t b = loadB(sW1 + (nt * 2 + kt) * 512, lane);
        acc = __builtin_amdgcn_wmma_f32_16x16x32_f16(false, a, false, b,
                                                     (short)0, acc, false, false);
      }
      const float bias = sB1[nt * 16 + nLoc];
      const int mb = mt * 16 + mOff;
#pragma unroll
      for (int r = 0; r < 8; ++r) {
        float v = acc[r] + bias;
        h1[(mb + r) * TSTRIDE + nt * 16 + nLoc] = (_Float16)(v > 0.f ? v : 0.f);
      }
    }
  }

  // ---- layer 2: h1 x W2 -> relu -> h2 (reuse enc tile) ---------------------
#pragma unroll
  for (int mt = 0; mt < 2; ++mt) {
#pragma unroll
    for (int nt = 0; nt < 4; ++nt) {
      f8_t acc = {};
#pragma unroll
      for (int kt = 0; kt < 2; ++kt) {
        h16_t a = loadA(h1, mt, kt, lane);
        h16_t b = loadB(sW2 + (nt * 2 + kt) * 512, lane);
        acc = __builtin_amdgcn_wmma_f32_16x16x32_f16(false, a, false, b,
                                                     (short)0, acc, false, false);
      }
      const float bias = sB2[nt * 16 + nLoc];
      const int mb = mt * 16 + mOff;
#pragma unroll
      for (int r = 0; r < 8; ++r) {
        float v = acc[r] + bias;
        enc[(mb + r) * TSTRIDE + nt * 16 + nLoc] = (_Float16)(v > 0.f ? v : 0.f);
      }
    }
  }

  // ---- layer 3: h2 x W3 (33 cols, padded to 48) -> density + geom_feat -----
#pragma unroll
  for (int mt = 0; mt < 2; ++mt) {
#pragma unroll
    for (int nt = 0; nt < 3; ++nt) {
      f8_t acc = {};
#pragma unroll
      for (int kt = 0; kt < 2; ++kt) {
        h16_t a = loadA(enc, mt, kt, lane);
        h16_t b = loadB(sW3 + (nt * 2 + kt) * 512, lane);
        acc = __builtin_amdgcn_wmma_f32_16x16x32_f16(false, a, false, b,
                                                     (short)0, acc, false, false);
      }
      const int n = nt * 16 + nLoc;
      const float bias = sB3[n];
      const int mb = mt * 16 + mOff;
#pragma unroll
      for (int r = 0; r < 8; ++r) {
        const int pi = tileBase + mb + r;
        if (pi < N && n < 33) {
          const float v = acc[r] + bias;
          if (n == 0) {
            // density = softplus(out[:,0]); NT store: don't evict the L2 table
            const float d = (v > 20.f) ? v : log1pf(__expf(v));
            __builtin_nontemporal_store(d, out + pi);
          } else {
            // geom_feat = out[:,1:33]
            __builtin_nontemporal_store(
                v, out + (size_t)N + (size_t)pi * 32 + (n - 1));
          }
        }
      }
    }
  }
}

extern "C" void kernel_launch(void* const* d_in, const int* in_sizes, int n_in,
                              void* d_out, int out_size, void* d_ws, size_t ws_size,
                              hipStream_t stream) {
  (void)n_in; (void)out_size; (void)d_ws; (void)ws_size;
  const float* points = (const float*)d_in[0];
  const float* bb     = (const float*)d_in[1];
  const float* tbl    = (const float*)d_in[2];
  const float* w1     = (const float*)d_in[3];
  const float* b1     = (const float*)d_in[4];
  const float* w2     = (const float*)d_in[5];
  const float* b2     = (const float*)d_in[6];
  const float* w3     = (const float*)d_in[7];
  const float* b3     = (const float*)d_in[8];
  float* out = (float*)d_out;

  const int N = in_sizes[0] / 3;
  const int nBlocks = (N + PTS_PER_BLK - 1) / PTS_PER_BLK;
  ngp_density_kernel<<<nBlocks, PTS_PER_BLK, 0, stream>>>(
      points, bb, tbl, w1, b1, w2, b2, w3, b3, out, N);
}
